// SpatialAttentionBlock_26714696581163
// MI455X (gfx1250) — matrix operand order repaired and checked by reading, not compiled, not run on hardware
//
#include <hip/hip_runtime.h>
#include <hip/hip_bf16.h>

typedef __attribute__((ext_vector_type(16))) __bf16 v16bf;
typedef __attribute__((ext_vector_type(8)))  float  v8f;

union BfReg { unsigned int u[8]; v16bf v; };

__device__ __forceinline__ unsigned short f2bf(float f) {
  unsigned int u = __float_as_uint(f);
  u += 0x7FFFu + ((u >> 16) & 1u);           // round-to-nearest-even
  return (unsigned short)(u >> 16);
}
__device__ __forceinline__ float bf2f(unsigned short h) {
  return __uint_as_float(((unsigned int)h) << 16);
}
__device__ __forceinline__ v8f wmma_bf16(const BfReg& a, const BfReg& b, v8f c) {
  return __builtin_amdgcn_wmma_f32_16x16x32_bf16(false, a.v, false, b.v,
                                                 (short)0, c, false, false);
}
__device__ __forceinline__ void fill8(BfReg& d, uint4 lo, uint4 hi) {
  d.u[0] = lo.x; d.u[1] = lo.y; d.u[2] = lo.z; d.u[3] = lo.w;
  d.u[4] = hi.x; d.u[5] = hi.y; d.u[6] = hi.z; d.u[7] = hi.w;
}

// ---------------------------------------------------------------------------
// B pack layout ("register order"): for K-step s (32 K-values), lane column n,
// lane half: 8 consecutive uints hold B VGPRs v=0..7 with k2 = 16*s+4*half+(v&3)+8*(v>>2).
//   BP2[(((size_t)s*N + n)*2 + half)*8 + v]
// A VGPR layout (16-bit A 16x32): v<4 -> K=8*half+2v, v>=4 -> K=16+8*half+2(v-4)
// => two contiguous b128 loads per row per K-step.
// ---------------------------------------------------------------------------
// 32x64 WMMA tile: rows m0..m0+31, cols n0..n0+63.
// acc[2*t+i]: N-subtile t, M-subtile i.
__device__ __forceinline__ void gemm_tile_32x64(
    const unsigned short* __restrict__ A, const unsigned int* __restrict__ Bp2,
    int m0, int n0, int K, int N, int lane, v8f acc[8]) {
  const int lc = lane & 15, half = lane >> 4;
  const unsigned short* arow0 = A + (size_t)(m0 + lc) * K;
  const unsigned short* arow1 = A + (size_t)(m0 + 16 + lc) * K;
  for (int kk = 0; kk < K; kk += 32) {
    BfReg a0, a1;
    fill8(a0, *(const uint4*)(arow0 + kk + 8 * half),
              *(const uint4*)(arow0 + kk + 16 + 8 * half));
    fill8(a1, *(const uint4*)(arow1 + kk + 8 * half),
              *(const uint4*)(arow1 + kk + 16 + 8 * half));
    const int s = kk >> 5;
#pragma unroll
    for (int t = 0; t < 4; ++t) {
      const uint4* bp = (const uint4*)(
          Bp2 + (((size_t)s * N + n0 + 16 * t + lc) * 2 + half) * 8);
      BfReg b;
      fill8(b, bp[0], bp[1]);
      acc[2 * t]     = wmma_bf16(a0, b, acc[2 * t]);
      acc[2 * t + 1] = wmma_bf16(a1, b, acc[2 * t + 1]);
    }
  }
}

// ---------------------------------------------------------------------------
// Prep kernels
// ---------------------------------------------------------------------------
__global__ void cvt_bf16_kernel(const float* __restrict__ x,
                                unsigned short* __restrict__ y, int n) {
  int i = blockIdx.x * blockDim.x + threadIdx.x;
  if (i < n) y[i] = f2bf(x[i]);
}

// Pack W[K][N] fp32 -> register-order bf16-pair layout (see above)
__global__ void pack_w_kernel(const float* __restrict__ W,
                              unsigned int* __restrict__ Wp, int K, int N) {
  int i = blockIdx.x * blockDim.x + threadIdx.x;
  int total = (K >> 1) * N;
  if (i >= total) return;
  int v = i & 7;
  int half = (i >> 3) & 1;
  int rest = i >> 4;            // [0, (K/32)*N)
  int n = rest % N;
  int s = rest / N;
  int k2 = 16 * s + 4 * half + (v & 3) + 8 * (v >> 2);
  unsigned int lo = f2bf(W[(size_t)(2 * k2) * N + n]);
  unsigned int hi = f2bf(W[(size_t)(2 * k2 + 1) * N + n]);
  Wp[i] = lo | (hi << 16);
}

// ---------------------------------------------------------------------------
// QKV GEMM: [8192,512] x [512,1536] -> scatter into per-head bf16 Q/K/V
// grid (24, 64), block 128 (4 waves, each one 32x64 tile)
// ---------------------------------------------------------------------------
__global__ __launch_bounds__(128) void gemm_qkv_kernel(
    const unsigned short* __restrict__ Aq, const unsigned int* __restrict__ Bp,
    const float* __restrict__ bias, unsigned short* __restrict__ qbf,
    unsigned short* __restrict__ kbf, unsigned short* __restrict__ vbf) {
  const int wave = threadIdx.x >> 5, lane = threadIdx.x & 31;
  const int lc = lane & 15, half = lane >> 4;
  const int m0 = (blockIdx.y * 4 + wave) * 32;
  const int n0 = blockIdx.x * 64;
  v8f acc[8];
#pragma unroll
  for (int t = 0; t < 8; ++t)
#pragma unroll
    for (int r = 0; r < 8; ++r) acc[t][r] = 0.f;
  gemm_tile_32x64(Aq, Bp, m0, n0, 512, 1536, lane, acc);
#pragma unroll
  for (int i = 0; i < 2; ++i) {
#pragma unroll
    for (int t = 0; t < 4; ++t) {
#pragma unroll
      for (int r = 0; r < 8; ++r) {
        int m = m0 + 16 * i + r + 8 * half;
        int col = n0 + 16 * t + lc;
        float val = acc[2 * t + i][r] + bias[col];
        int b = m >> 10, q = m & 1023;
        int g = col >> 6, h = col & 63;
        unsigned short bv = f2bf(val);
        if (g < 8)
          qbf[((size_t)(b * 8 + g) * 1024 + q) * 64 + h] = bv;
        else if (g < 16)
          kbf[((size_t)(b * 8 + (g - 8)) * 1024 + q) * 64 + h] = bv;
        else
          vbf[((size_t)(b * 8 + (g - 16)) * 1024 + q) * 64 + h] = bv;
      }
    }
  }
}

// ---------------------------------------------------------------------------
// Locator: offsets/factors per (b,n,q)
// ---------------------------------------------------------------------------
__global__ void locator_kernel(const unsigned short* __restrict__ qbf,
                               const float* __restrict__ locations,
                               const float* __restrict__ Woff, const float* __restrict__ boff,
                               const float* __restrict__ Wfac, const float* __restrict__ bfac,
                               float* __restrict__ locs, float* __restrict__ facs) {
  int i = blockIdx.x * blockDim.x + threadIdx.x;
  if (i >= 64 * 1024) return;
  int bn = i >> 10, q = i & 1023;
  int b = bn >> 3;
  const unsigned short* row = qbf + (size_t)i * 64;
  float o0 = boff[0], o1 = boff[1], a0 = bfac[0], a1 = bfac[1];
#pragma unroll 8
  for (int h = 0; h < 64; ++h) {
    float x = bf2f(row[h]);
    o0 += x * Woff[h * 2 + 0];
    o1 += x * Woff[h * 2 + 1];
    a0 += x * Wfac[h * 2 + 0];
    a1 += x * Wfac[h * 2 + 1];
  }
  a0 = (a0 > 20.f) ? a0 : log1pf(__expf(a0));
  a1 = (a1 > 20.f) ? a1 : log1pf(__expf(a1));
  float lx = locations[((size_t)b * 1024 + q) * 2 + 0] + o0;
  float ly = locations[((size_t)b * 1024 + q) * 2 + 1] + o1;
  locs[(size_t)i * 2 + 0] = lx;
  locs[(size_t)i * 2 + 1] = ly;
  facs[(size_t)i * 2 + 0] = a0;
  facs[(size_t)i * 2 + 1] = a1;
}

// ---------------------------------------------------------------------------
// Pack K for score-B (register order): per bn, 2 K-steps (h2 16 each), N=1024.
//   KP[bn][((s*1024 + v)*2 + half)*8 + r],  h2 = 16*s + 4*half + (r&3) + 8*(r>>2)
// ---------------------------------------------------------------------------
__global__ void pack_k_kernel(const unsigned short* __restrict__ kbf,
                              unsigned int* __restrict__ kpack) {
  int i = blockIdx.x * blockDim.x + threadIdx.x;
  if (i >= 64 * 32 * 1024) return;
  int bn = i >> 15;
  int j = i & 32767;
  int r = j & 7;
  int half = (j >> 3) & 1;
  int rest = j >> 4;            // 0..2047
  int v = rest & 1023;
  int s = rest >> 10;           // 0..1
  int h2 = 16 * s + 4 * half + (r & 3) + 8 * (r >> 2);
  unsigned int lo = kbf[((size_t)bn * 1024 + v) * 64 + 2 * h2];
  unsigned int hi = kbf[((size_t)bn * 1024 + v) * 64 + 2 * h2 + 1];
  kpack[i] = lo | (hi << 16);
}

// Pack V for PV-B (register order): per bn, 32 K-steps over v (v2 16 each), N=64.
//   VP[bn][((sv*64 + h)*2 + half)*8 + r],  v2 = 16*sv + 4*half + (r&3) + 8*(r>>2)
__global__ void pack_v_kernel(const unsigned short* __restrict__ vbf,
                              unsigned int* __restrict__ vpack) {
  int i = blockIdx.x * blockDim.x + threadIdx.x;
  if (i >= 64 * 512 * 64) return;
  int bn = i >> 15;
  int j = i & 32767;
  int r = j & 7;
  int half = (j >> 3) & 1;
  int rest = j >> 4;            // 0..2047
  int h = rest & 63;
  int sv = rest >> 6;           // 0..31
  int v2 = 16 * sv + 4 * half + (r & 3) + 8 * (r >> 2);
  unsigned int lo = vbf[((size_t)bn * 1024 + 2 * v2) * 64 + h];
  unsigned int hi = vbf[((size_t)bn * 1024 + 2 * v2 + 1) * 64 + h];
  vpack[i] = lo | (hi << 16);
}

// ---------------------------------------------------------------------------
// Fused flash attention with spatial Gaussian log-weights.
// grid (16, 64), block 128.  Wave handles 16 q-rows; v-loop in tiles of 32.
// ---------------------------------------------------------------------------
__global__ __launch_bounds__(128) void flash_attn_kernel(
    const unsigned short* __restrict__ qbf, const unsigned int* __restrict__ kpack,
    const unsigned int* __restrict__ vpack, const float* __restrict__ locs,
    const float* __restrict__ facs, const float* __restrict__ locations,
    const unsigned char* __restrict__ masks, unsigned short* __restrict__ ohbf) {
  __shared__ __align__(16) unsigned short lds[4][16][32];
  const int wave = threadIdx.x >> 5, lane = threadIdx.x & 31;
  const int lc = lane & 15, half = lane >> 4;
  const int bn = blockIdx.y;
  const int b = bn >> 3, n = bn & 7;
  const int q0 = (blockIdx.x * 4 + wave) * 16;

  // A-tiles for Q (16x64 -> two 16x32 K-steps)
  const unsigned short* arow = qbf + ((size_t)bn * 1024 + q0 + lc) * 64;
  BfReg A0, A1;
  fill8(A0, *(const uint4*)(arow + 8 * half), *(const uint4*)(arow + 16 + 8 * half));
  fill8(A1, *(const uint4*)(arow + 32 + 8 * half), *(const uint4*)(arow + 48 + 8 * half));

  // per-row locator params (C-layout rows: m = r + 8*half)
  float lx[8], ly[8], f0[8], f1[8], rmax[8], rsum[8];
#pragma unroll
  for (int r = 0; r < 8; ++r) {
    int qq = q0 + r + 8 * half;
    size_t idx = ((size_t)bn * 1024 + qq) * 2;
    lx[r] = locs[idx]; ly[r] = locs[idx + 1];
    f0[r] = facs[idx]; f1[r] = facs[idx + 1];
    rmax[r] = -1e30f; rsum[r] = 0.f;
  }
  v8f accO[4];
#pragma unroll
  for (int t = 0; t < 4; ++t)
#pragma unroll
    for (int r = 0; r < 8; ++r) accO[t][r] = 0.f;

  const unsigned int* kb = kpack + (size_t)bn * 32 * 1024;
  const unsigned int* vb = vpack + (size_t)bn * 512 * 64;

  for (int vt = 0; vt < 1024; vt += 32) {
    // ---- score: two 16x16 subtiles, K=64 in two WMMA steps ----
    v8f S[2];
#pragma unroll
    for (int c = 0; c < 2; ++c) {
#pragma unroll
      for (int r = 0; r < 8; ++r) S[c][r] = 0.f;
      const int vcol = vt + 16 * c + lc;
      const uint4* bp0 = (const uint4*)(kb + (((size_t)0 * 1024 + vcol) * 2 + half) * 8);
      const uint4* bp1 = (const uint4*)(kb + (((size_t)1 * 1024 + vcol) * 2 + half) * 8);
      BfReg B0, B1;
      fill8(B0, bp0[0], bp0[1]);
      fill8(B1, bp1[0], bp1[1]);
      S[c] = wmma_bf16(A0, B0, S[c]);
      S[c] = wmma_bf16(A1, B1, S[c]);
    }
    // ---- logits = score/8 + gaussian log-weight, masked ----
    float P[2][8];
#pragma unroll
    for (int c = 0; c < 2; ++c) {
      int v = vt + 16 * c + lc;
      float kx = locations[((size_t)b * 1024 + v) * 2 + 0];
      float ky = locations[((size_t)b * 1024 + v) * 2 + 1];
      bool mk = masks[(size_t)b * 1024 + v] != 0;
#pragma unroll
      for (int r = 0; r < 8; ++r) {
        float dx = kx - lx[r], dy = ky - ly[r];
        float val = S[c][r] * 0.125f - 0.5f * (f0[r] * dx * dx + f1[r] * dy * dy);
        P[c][r] = mk ? val : -1e9f;
      }
    }
    // ---- online softmax (row = 16 lanes in a half) ----
    float e0[8], e1[8], corr[8];
#pragma unroll
    for (int r = 0; r < 8; ++r) {
      float m = fmaxf(P[0][r], P[1][r]);
      for (int o = 8; o >= 1; o >>= 1) m = fmaxf(m, __shfl_xor(m, o, 32));
      float nm = fmaxf(rmax[r], m);
      corr[r] = __expf(rmax[r] - nm);
      rmax[r] = nm;
      e0[r] = __expf(P[0][r] - nm);
      e1[r] = __expf(P[1][r] - nm);
      float s = e0[r] + e1[r];
      for (int o = 8; o >= 1; o >>= 1) s += __shfl_xor(s, o, 32);
      rsum[r] = rsum[r] * corr[r] + s;
    }
#pragma unroll
    for (int t = 0; t < 4; ++t)
#pragma unroll
      for (int r = 0; r < 8; ++r) accO[t][r] *= corr[r];

    // ---- C-layout -> A-layout re-striping through LDS (bf16 probs) ----
    __syncthreads();
#pragma unroll
    for (int r = 0; r < 8; ++r) {
      lds[wave][r + 8 * half][lc] = f2bf(e0[r]);
      lds[wave][r + 8 * half][16 + lc] = f2bf(e1[r]);
    }
    __syncthreads();
    BfReg PA;
    fill8(PA, *(const uint4*)&lds[wave][lc][8 * half],
              *(const uint4*)&lds[wave][lc][16 + 8 * half]);
    // ---- PV: 4 h-subtiles ----
    const int sv = vt >> 5;
#pragma unroll
    for (int t = 0; t < 4; ++t) {
      const uint4* vp = (const uint4*)(
          vb + (((size_t)sv * 64 + 16 * t + lc) * 2 + half) * 8);
      BfReg BV;
      fill8(BV, vp[0], vp[1]);
      accO[t] = wmma_bf16(PA, BV, accO[t]);
    }
  }
  // ---- finalize: /rowsum, store outputs_h as bf16 [b][q][n][h] ----
  float rinv[8];
#pragma unroll
  for (int r = 0; r < 8; ++r) rinv[r] = 1.f / rsum[r];
#pragma unroll
  for (int t = 0; t < 4; ++t) {
#pragma unroll
    for (int r = 0; r < 8; ++r) {
      int qq = q0 + r + 8 * half;
      ohbf[(((size_t)b * 1024 + qq) * 8 + n) * 64 + 16 * t + lc] =
          f2bf(accO[t][r] * rinv[r]);
    }
  }
}

// ---------------------------------------------------------------------------
// Anchor MLP + head softmax -> locations_out
// ---------------------------------------------------------------------------
__global__ void anchor_kernel(const unsigned short* __restrict__ ohbf,
                              const float* __restrict__ locs,
                              const float* __restrict__ Wa1, const float* __restrict__ ba1,
                              const float* __restrict__ Wa2, const float* __restrict__ ba2,
                              float* __restrict__ locout) {
  int i = blockIdx.x * blockDim.x + threadIdx.x;
  if (i >= 8192) return;
  int b = i >> 10, q = i & 1023;
  float hs[8][2];
  for (int n = 0; n < 8; ++n) {
    const unsigned short* row = ohbf + ((size_t)i * 8 + n) * 64;
    float h1[32];
    for (int j = 0; j < 32; ++j) {
      float s = ba1[j];
      for (int h = 0; h < 64; ++h) s += bf2f(row[h]) * Wa1[h * 32 + j];
      h1[j] = 0.5f * s * (1.f + erff(s * 0.70710678f));   // exact GELU
    }
    for (int c = 0; c < 2; ++c) {
      float s = ba2[c];
      for (int j = 0; j < 32; ++j) s += h1[j] * Wa2[j * 2 + c];
      hs[n][c] = s;
    }
  }
  for (int c = 0; c < 2; ++c) {
    float m = -1e30f;
    for (int n = 0; n < 8; ++n) m = fmaxf(m, hs[n][c]);
    float s = 0.f;
    float e[8];
    for (int n = 0; n < 8; ++n) { e[n] = __expf(hs[n][c] - m); s += e[n]; }
    float acc = 0.f;
    for (int n = 0; n < 8; ++n)
      acc += (e[n] / s) * locs[(((size_t)(b * 8 + n)) * 1024 + q) * 2 + c];
    locout[(size_t)i * 2 + c] = acc;
  }
}

// ---------------------------------------------------------------------------
// Output GEMM: [8192,512] x [512,512] + bias + residual -> outpre
// grid (8, 64), block 128
// ---------------------------------------------------------------------------
__global__ __launch_bounds__(128) void gemm_out_kernel(
    const unsigned short* __restrict__ Aoh, const unsigned int* __restrict__ Bp,
    const float* __restrict__ bias, const float* __restrict__ queries,
    float* __restrict__ outpre) {
  const int wave = threadIdx.x >> 5, lane = threadIdx.x & 31;
  const int lc = lane & 15, half = lane >> 4;
  const int m0 = (blockIdx.y * 4 + wave) * 32;
  const int n0 = blockIdx.x * 64;
  v8f acc[8];
#pragma unroll
  for (int t = 0; t < 8; ++t)
#pragma unroll
    for (int r = 0; r < 8; ++r) acc[t][r] = 0.f;
  gemm_tile_32x64(Aoh, Bp, m0, n0, 512, 512, lane, acc);
#pragma unroll
  for (int i = 0; i < 2; ++i) {
#pragma unroll
    for (int t = 0; t < 4; ++t) {
#pragma unroll
      for (int r = 0; r < 8; ++r) {
        int m = m0 + 16 * i + r + 8 * half;
        int col = n0 + 16 * t + lc;
        outpre[(size_t)m * 512 + col] =
            acc[2 * t + i][r] + bias[col] + queries[(size_t)m * 512 + col];
      }
    }
  }
}

// ---------------------------------------------------------------------------
// LayerNorm per row, one wave per row
// ---------------------------------------------------------------------------
__global__ __launch_bounds__(128) void layernorm_kernel(
    const float* __restrict__ outpre, const float* __restrict__ gamma,
    const float* __restrict__ beta, float* __restrict__ dst) {
  const int wave = threadIdx.x >> 5, lane = threadIdx.x & 31;
  const int row = blockIdx.x * 4 + wave;
  const float* x = outpre + (size_t)row * 512;
  float s = 0.f, ss = 0.f;
  for (int j = lane; j < 512; j += 32) {
    float v = x[j];
    s += v; ss += v * v;
  }
  for (int o = 16; o >= 1; o >>= 1) {
    s += __shfl_xor(s, o, 32);
    ss += __shfl_xor(ss, o, 32);
  }
  float mu = s * (1.f / 512.f);
  float var = ss * (1.f / 512.f) - mu * mu;
  float inv = rsqrtf(var + 1e-5f);
  for (int j = lane; j < 512; j += 32)
    dst[(size_t)row * 512 + j] = (x[j] - mu) * inv * gamma[j] + beta[j];
}

// ---------------------------------------------------------------------------
extern "C" void kernel_launch(void* const* d_in, const int* in_sizes, int n_in,
                              void* d_out, int out_size, void* d_ws, size_t ws_size,
                              hipStream_t stream) {
  const float* queries   = (const float*)d_in[0];
  const float* locations = (const float*)d_in[1];
  const unsigned char* masks = (const unsigned char*)d_in[2];
  const float* W_qkv = (const float*)d_in[3];
  const float* b_qkv = (const float*)d_in[4];
  const float* W_off = (const float*)d_in[5];
  const float* b_off = (const float*)d_in[6];
  const float* W_fac = (const float*)d_in[7];
  const float* b_fac = (const float*)d_in[8];
  const float* W_a1  = (const float*)d_in[9];
  const float* b_a1  = (const float*)d_in[10];
  const float* W_a2  = (const float*)d_in[11];
  const float* b_a2  = (const float*)d_in[12];
  const float* W_o   = (const float*)d_in[13];
  const float* b_o   = (const float*)d_in[14];
  const float* gamma = (const float*)d_in[15];
  const float* beta  = (const float*)d_in[16];

  float* out_q   = (float*)d_out;                           // [8,1024,512]
  float* out_loc = (float*)d_out + (size_t)8 * 1024 * 512;  // [8,1024,2]

  unsigned char* W = (unsigned char*)d_ws;
  size_t off = 0;
  auto give = [&](size_t bytes) {
    size_t o = off;
    off += (bytes + 255) & ~(size_t)255;
    return o;
  };
  unsigned short* q_in_bf = (unsigned short*)(W + give((size_t)8192 * 512 * 2));
  unsigned int*   wqkv_p  = (unsigned int*)  (W + give((size_t)256 * 1536 * 4));
  unsigned int*   wo_p    = (unsigned int*)  (W + give((size_t)256 * 512 * 4));
  unsigned short* qbf     = (unsigned short*)(W + give((size_t)64 * 1024 * 64 * 2));
  unsigned short* kbf     = (unsigned short*)(W + give((size_t)64 * 1024 * 64 * 2));
  unsigned short* vbf     = (unsigned short*)(W + give((size_t)64 * 1024 * 64 * 2));
  unsigned int*   kpk     = (unsigned int*)  (W + give((size_t)64 * 32 * 1024 * 4));
  unsigned int*   vpk     = (unsigned int*)  (W + give((size_t)64 * 512 * 64 * 4));
  float*          locs    = (float*)         (W + give((size_t)64 * 1024 * 2 * 4));
  float*          facs    = (float*)         (W + give((size_t)64 * 1024 * 2 * 4));
  unsigned short* ohbf    = (unsigned short*)(W + give((size_t)8192 * 512 * 2));
  float*          outpre  = (float*)         (W + give((size_t)8192 * 512 * 4));

  // 1) conversions / weight packing
  cvt_bf16_kernel<<<(8192 * 512 + 255) / 256, 256, 0, stream>>>(
      queries, q_in_bf, 8192 * 512);
  pack_w_kernel<<<(256 * 1536 + 255) / 256, 256, 0, stream>>>(W_qkv, wqkv_p, 512, 1536);
  pack_w_kernel<<<(256 * 512 + 255) / 256, 256, 0, stream>>>(W_o, wo_p, 512, 512);

  // 2) QKV GEMM (WMMA)
  gemm_qkv_kernel<<<dim3(24, 64), 128, 0, stream>>>(q_in_bf, wqkv_p, b_qkv,
                                                    qbf, kbf, vbf);
  // 3) locator
  locator_kernel<<<(64 * 1024 + 255) / 256, 256, 0, stream>>>(
      qbf, locations, W_off, b_off, W_fac, b_fac, locs, facs);
  // 4) K/V packing for attention WMMA B layouts
  pack_k_kernel<<<(64 * 32 * 1024 + 255) / 256, 256, 0, stream>>>(kbf, kpk);
  pack_v_kernel<<<(64 * 512 * 64 + 255) / 256, 256, 0, stream>>>(vbf, vpk);
  // 5) fused flash attention (WMMA)
  flash_attn_kernel<<<dim3(16, 64), 128, 0, stream>>>(qbf, kpk, vpk, locs, facs,
                                                      locations, masks, ohbf);
  // 6) anchor MLP -> locations_out
  anchor_kernel<<<(8192 + 127) / 128, 128, 0, stream>>>(ohbf, locs, W_a1, b_a1,
                                                        W_a2, b_a2, out_loc);
  // 7) output GEMM (WMMA) + residual
  gemm_out_kernel<<<dim3(8, 64), 128, 0, stream>>>(ohbf, wo_p, b_o, queries, outpre);
  // 8) LayerNorm -> queries_out
  layernorm_kernel<<<2048, 128, 0, stream>>>(outpre, gamma, beta, out_q);
}
